// CausalSelfAttention_38328288150352
// MI455X (gfx1250) — matrix operand order repaired and checked by reading, not compiled, not run on hardware
//
#include <hip/hip_runtime.h>
#include <cstdint>
#include <cstddef>

// ---------------------------------------------------------------------------
// CDNA5 (gfx1250) causal self-attention forward.
// bf16 WMMA (v_wmma_f32_16x16x32_bf16) everywhere, fp32 accumulation,
// flash-attention inner loop, double-buffered LDS staging with (when the
// toolchain exposes it) GLOBAL_LOAD_ASYNC_TO_LDS_B128 + s_wait_asynccnt.
// ---------------------------------------------------------------------------

typedef __bf16 bf16;
typedef __bf16 v16bf  __attribute__((ext_vector_type(16)));
typedef __bf16 bf16x8 __attribute__((ext_vector_type(8)));
typedef float  v8f    __attribute__((ext_vector_type(8)));

constexpr int kB  = 2;
constexpr int kT  = 2048;
constexpr int kC  = 2048;
constexpr int kNH = 16;
constexpr int kHD = 128;
constexpr int kN3 = 3 * kC;

// LDS strides (bf16 elements) padded so per-lane ds_load_b128 fragment
// fetches hit 16 disjoint 4-bank groups (64 banks).
constexpr int LDT = 40;   // 32-deep K tiles (32 + 8 pad)
constexpr int LKV = 136;  // 128-deep K tiles (128 + 8 pad)

#define LOG2E 1.4426950408889634f

// ---- async global->LDS path (gfx1250), guarded for toolchain support ------
#if defined(__has_builtin)
#if __has_builtin(__builtin_amdgcn_global_load_async_to_lds_b128) && \
    __has_builtin(__builtin_amdgcn_s_wait_asynccnt)
#define USE_ASYNC_LDS 1
#endif
#endif

#ifdef USE_ASYNC_LDS
// Builtin signature (from clang diagnostic): int4 in addrspace(1) source,
// int4 in addrspace(3) LDS destination, imm offset, imm cpol.
typedef int v4i __attribute__((ext_vector_type(4)));
typedef __attribute__((address_space(1))) v4i gv4i;
typedef __attribute__((address_space(3))) v4i lv4i;
__device__ __forceinline__ void cp16_async(void* ldsDst, const void* gSrc) {
  __builtin_amdgcn_global_load_async_to_lds_b128((gv4i*)gSrc, (lv4i*)ldsDst, 0, 0);
}
__device__ __forceinline__ void wait_async() {
  __builtin_amdgcn_s_wait_asynccnt(0);
}
#endif

__device__ __forceinline__ v8f zero8() {
  v8f z;
#pragma unroll
  for (int i = 0; i < 8; ++i) z[i] = 0.0f;
  return z;
}

__device__ __forceinline__ v16bf pack16(bf16x8 lo, bf16x8 hi) {
  v16bf r;
#pragma unroll
  for (int i = 0; i < 8; ++i) { r[i] = lo[i]; r[i + 8] = hi[i]; }
  return r;
}

// D = A(16x32) * B(32x16) + C, bf16 in, f32 accum.
__device__ __forceinline__ v8f wmma_bf16(v16bf a, v16bf b, v8f c) {
  return __builtin_amdgcn_wmma_f32_16x16x32_bf16(
      false, a, false, b, (short)0, c, false, false);
}

// A-fragment (16x32, 16-bit): lane<16 -> row=lane, K 0..7 & 16..23;
// lane>=16 -> row=lane-16, K 8..15 & 24..31.  src row-major, stride lds.
__device__ __forceinline__ v16bf load_afrag(const bf16* src, int lds, int l16, int hi) {
  const bf16* p = src + l16 * lds + hi * 8;
  return pack16(*(const bf16x8*)(p), *(const bf16x8*)(p + 16));
}

// B-fragment (32x16, 16-bit) from N-major storage [n][k]:
// lane -> n = lane&15; lane<16 -> K 0..7 & 16..23, lane>=16 -> K 8..15 & 24..31.
__device__ __forceinline__ v16bf load_bfrag(const bf16* srcNmajor, int lds, int l16, int hi) {
  const bf16* p = srcNmajor + l16 * lds + hi * 8;
  return pack16(*(const bf16x8*)(p), *(const bf16x8*)(p + 16));
}

// ---------------------------------------------------------------------------
// Kernel 1: fp32 -> bf16 elementwise
// ---------------------------------------------------------------------------
__global__ void f32_to_bf16_kernel(const float* __restrict__ src,
                                   bf16* __restrict__ dst, size_t n) {
  size_t i = (size_t)blockIdx.x * blockDim.x + threadIdx.x;
  size_t stride = (size_t)gridDim.x * blockDim.x;
  for (; i < n; i += stride) dst[i] = (bf16)src[i];
}

// ---------------------------------------------------------------------------
// Kernel 2: fused transpose + fp32->bf16. src: K x N row-major (fp32),
// dst: N x K row-major (bf16) -> N-major B operand for WMMA.
// ---------------------------------------------------------------------------
__global__ __launch_bounds__(256) void transpose_to_bf16_kernel(
    const float* __restrict__ src, bf16* __restrict__ dst, int K, int N) {
  __shared__ float tile[32][33];
  const int k0 = blockIdx.y * 32;
  const int n0 = blockIdx.x * 32;
  const int tx = threadIdx.x & 31;
  const int ty = threadIdx.x >> 5;
#pragma unroll
  for (int i = 0; i < 32; i += 8)
    tile[ty + i][tx] = src[(size_t)(k0 + ty + i) * N + n0 + tx];
  __syncthreads();
#pragma unroll
  for (int i = 0; i < 32; i += 8)
    dst[(size_t)(n0 + ty + i) * K + k0 + tx] = (bf16)tile[tx][ty + i];
}

// ---------------------------------------------------------------------------
// Kernel 3: QKV GEMM (M=4096, N=6144, K=2048), double-buffered LDS.
// Block tile 128x128, 8 waves as 4(M) x 2(N), wave tile 32x64, K step 32.
// Epilogue adds bias and scatters to q (pre-scaled), k, v^T (bf16).
// ---------------------------------------------------------------------------
__global__ __launch_bounds__(256) void qkv_gemm_kernel(
    const bf16* __restrict__ xb,     // (B*T, C) row-major
    const bf16* __restrict__ wT,     // (3C, C)  N-major
    const float* __restrict__ bias,  // (3C)
    bf16* __restrict__ qO, bf16* __restrict__ kO, bf16* __restrict__ vtO) {
  __shared__ __align__(16) bf16 aLds[2][128 * LDT];
  __shared__ __align__(16) bf16 bLds[2][128 * LDT];

  const int tid  = threadIdx.x;
  const int wave = tid >> 5;
  const int lane = tid & 31;
  const int l16  = lane & 15;
  const int hi   = lane >> 4;
  const int waveM = wave & 3;
  const int waveN = wave >> 2;
  const int mBlk = blockIdx.y * 128;
  const int nBlk = blockIdx.x * 128;

  v8f acc[2][4];
#pragma unroll
  for (int mt = 0; mt < 2; ++mt)
#pragma unroll
    for (int nt = 0; nt < 4; ++nt) acc[mt][nt] = zero8();

  const int r   = tid >> 1;        // 0..127
  const int seg = (tid & 1) * 16;  // 0 / 16
  const int off = r * LDT + seg;
  const bf16* srcA = xb + (size_t)(mBlk + r) * kC + seg;
  const bf16* srcB = wT + (size_t)(nBlk + r) * kC + seg;

  // prologue: stage 0 into buffer 0
#ifdef USE_ASYNC_LDS
  cp16_async(&aLds[0][off], srcA);
  cp16_async(&aLds[0][off + 8], srcA + 8);
  cp16_async(&bLds[0][off], srcB);
  cp16_async(&bLds[0][off + 8], srcB + 8);
  wait_async();
#else
  *(bf16x8*)&aLds[0][off]     = *(const bf16x8*)(srcA);
  *(bf16x8*)&aLds[0][off + 8] = *(const bf16x8*)(srcA + 8);
  *(bf16x8*)&bLds[0][off]     = *(const bf16x8*)(srcB);
  *(bf16x8*)&bLds[0][off + 8] = *(const bf16x8*)(srcB + 8);
#endif
  __syncthreads();

  constexpr int S = kC / 32;  // 64 K-steps
  int buf = 0;
  for (int s = 0; s < S; ++s) {
    const int nbuf = buf ^ 1;
    const bool more = (s + 1) < S;
#ifdef USE_ASYNC_LDS
    if (more) {  // issue next stage straight into the other LDS buffer
      const bf16* na = srcA + (s + 1) * 32;
      const bf16* nb = srcB + (s + 1) * 32;
      cp16_async(&aLds[nbuf][off], na);
      cp16_async(&aLds[nbuf][off + 8], na + 8);
      cp16_async(&bLds[nbuf][off], nb);
      cp16_async(&bLds[nbuf][off + 8], nb + 8);
    }
#else
    bf16x8 ga0, ga1, gb0, gb1;
    if (more) {  // stage next tile in registers while we compute
      const bf16* na = srcA + (s + 1) * 32;
      const bf16* nb = srcB + (s + 1) * 32;
      ga0 = *(const bf16x8*)(na);
      ga1 = *(const bf16x8*)(na + 8);
      gb0 = *(const bf16x8*)(nb);
      gb1 = *(const bf16x8*)(nb + 8);
    }
#endif
    v16bf af[2], bf_[4];
#pragma unroll
    for (int mt = 0; mt < 2; ++mt)
      af[mt] = load_afrag(&aLds[buf][(waveM * 32 + mt * 16) * LDT], LDT, l16, hi);
#pragma unroll
    for (int nt = 0; nt < 4; ++nt)
      bf_[nt] = load_bfrag(&bLds[buf][(waveN * 64 + nt * 16) * LDT], LDT, l16, hi);
#pragma unroll
    for (int mt = 0; mt < 2; ++mt)
#pragma unroll
      for (int nt = 0; nt < 4; ++nt)
        acc[mt][nt] = wmma_bf16(af[mt], bf_[nt], acc[mt][nt]);

#ifdef USE_ASYNC_LDS
    if (more) wait_async();
#else
    if (more) {
      *(bf16x8*)&aLds[nbuf][off]     = ga0;
      *(bf16x8*)&aLds[nbuf][off + 8] = ga1;
      *(bf16x8*)&bLds[nbuf][off]     = gb0;
      *(bf16x8*)&bLds[nbuf][off + 8] = gb1;
    }
#endif
    __syncthreads();
    buf = nbuf;
  }

  const float qscale = 0.08838834764831845f;  // 1/sqrt(128)
#pragma unroll
  for (int mt = 0; mt < 2; ++mt) {
#pragma unroll
    for (int nt = 0; nt < 4; ++nt) {
#pragma unroll
      for (int v = 0; v < 8; ++v) {
        const uint32_t rowg = (uint32_t)(mBlk + waveM * 32 + mt * 16 + v + 8 * hi);
        const uint32_t colg = (uint32_t)(nBlk + waveN * 64 + nt * 16 + l16);
        const float val = acc[mt][nt][v] + bias[colg];
        const uint32_t which = colg >> 11;        // /2048 : 0=q 1=k 2=v
        const uint32_t c = colg & 2047u;
        const uint32_t h = c >> 7, d = c & 127u;  // /128, %128
        const uint32_t b = rowg >> 11, t = rowg & 2047u;
        const size_t bh = (size_t)(b * kNH + h);
        if (which == 0)      qO[(bh * kT + t) * kHD + d] = (bf16)(val * qscale);
        else if (which == 1) kO[(bh * kT + t) * kHD + d] = (bf16)val;
        else                 vtO[(bh * kHD + d) * kT + t] = (bf16)val;
      }
    }
  }
}

// ---------------------------------------------------------------------------
// Kernel 4: flash attention.  Block = 128 q rows of one (b,h); 8 waves, each
// owns 16 q rows with a 16x128 fp32 accumulator.  32-key tiles staged in
// double-buffered LDS; online softmax in fp32; P bounced through a per-wave
// LDS patch to convert C-layout -> A-layout for the PV WMMAs.
// ---------------------------------------------------------------------------
__global__ __launch_bounds__(256) void flash_attn_kernel(
    const bf16* __restrict__ q,   // (B,H,T,HD), pre-scaled
    const bf16* __restrict__ k,   // (B,H,T,HD)
    const bf16* __restrict__ vt,  // (B,H,HD,T)
    bf16* __restrict__ y) {       // (B,T,C)
  __shared__ __align__(16) bf16 kLds[2][32 * LKV];   // [key][d]
  __shared__ __align__(16) bf16 vLds[2][128 * LDT];  // [d][key]
  __shared__ __align__(16) bf16 pLds[8 * 16 * LDT];

  const int tid  = threadIdx.x;
  const int wave = tid >> 5;
  const int lane = tid & 31;
  const int l16  = lane & 15;
  const int hi   = lane >> 4;

  const uint32_t bh = blockIdx.y;
  const uint32_t b  = bh >> 4;   // / kNH
  const uint32_t h  = bh & 15u;  // % kNH
  const int qBase = blockIdx.x * 128;
  const int t0 = qBase + wave * 16;

  // Q fragments for this wave's 16 rows (K = HD = 128 -> 4 fragments).
  v16bf qf[4];
  {
    const bf16* qRow = q + ((size_t)bh * kT + t0 + l16) * kHD;
#pragma unroll
    for (int kc = 0; kc < 4; ++kc) {
      const bf16* p = qRow + kc * 32 + hi * 8;
      qf[kc] = pack16(*(const bf16x8*)(p), *(const bf16x8*)(p + 16));
    }
  }

  v8f yacc[8];
#pragma unroll
  for (int nt = 0; nt < 8; ++nt) yacc[nt] = zero8();
  float mi[8], li[8];
#pragma unroll
  for (int v = 0; v < 8; ++v) { mi[v] = -1e30f; li[v] = 0.0f; }

  bf16* pMy = &pLds[wave * 16 * LDT];

  // per-thread copy mapping for K (32x128) and V^T (128x32) tiles
  const int kr = tid >> 3, ks = (tid & 7) * 16;
  const int vd = tid >> 1, vh = (tid & 1) * 16;
  const int kOff = kr * LKV + ks;
  const int vOff = vd * LDT + vh;
  const bf16* kSrc0 = k  + ((size_t)bh * kT + kr) * kHD + ks;
  const bf16* vSrc0 = vt + ((size_t)bh * kHD + vd) * kT + vh;

  const int nkt = qBase / 32 + 4;  // key tiles covering keys <= qBase+127

  // prologue: stage key-tile 0 into buffer 0
#ifdef USE_ASYNC_LDS
  cp16_async(&kLds[0][kOff], kSrc0);
  cp16_async(&kLds[0][kOff + 8], kSrc0 + 8);
  cp16_async(&vLds[0][vOff], vSrc0);
  cp16_async(&vLds[0][vOff + 8], vSrc0 + 8);
  wait_async();
#else
  *(bf16x8*)&kLds[0][kOff]     = *(const bf16x8*)(kSrc0);
  *(bf16x8*)&kLds[0][kOff + 8] = *(const bf16x8*)(kSrc0 + 8);
  *(bf16x8*)&vLds[0][vOff]     = *(const bf16x8*)(vSrc0);
  *(bf16x8*)&vLds[0][vOff + 8] = *(const bf16x8*)(vSrc0 + 8);
#endif
  __syncthreads();

  int buf = 0;
  for (int kt = 0; kt < nkt; ++kt) {
    const int ktB = kt * 32;
    const int nbuf = buf ^ 1;
    const bool more = (kt + 1) < nkt;
#ifdef USE_ASYNC_LDS
    if (more) {
      const bf16* nk = kSrc0 + (size_t)(ktB + 32) * kHD;  // next 32 key rows
      const bf16* nv = vSrc0 + (ktB + 32);                // next 32 key cols
      cp16_async(&kLds[nbuf][kOff], nk);
      cp16_async(&kLds[nbuf][kOff + 8], nk + 8);
      cp16_async(&vLds[nbuf][vOff], nv);
      cp16_async(&vLds[nbuf][vOff + 8], nv + 8);
    }
#else
    bf16x8 gk0, gk1, gv0, gv1;
    if (more) {
      const bf16* nk = kSrc0 + (size_t)(ktB + 32) * kHD;
      const bf16* nv = vSrc0 + (ktB + 32);
      gk0 = *(const bf16x8*)(nk);
      gk1 = *(const bf16x8*)(nk + 8);
      gv0 = *(const bf16x8*)(nv);
      gv1 = *(const bf16x8*)(nv + 8);
    }
#endif

    // S = q . k^T for 2 key sub-tiles of 16 (K rows are N-major B operand).
    v8f s[2];
#pragma unroll
    for (int sub = 0; sub < 2; ++sub) {
      s[sub] = zero8();
#pragma unroll
      for (int kc = 0; kc < 4; ++kc) {
        v16bf bfrag = load_bfrag(&kLds[buf][sub * 16 * LKV + kc * 32], LKV, l16, hi);
        s[sub] = wmma_bf16(qf[kc], bfrag, s[sub]);
      }
    }

    // Causal mask in C-layout: M = v + 8*hi, N = l16.
#pragma unroll
    for (int sub = 0; sub < 2; ++sub)
#pragma unroll
      for (int v = 0; v < 8; ++v)
        if (ktB + sub * 16 + l16 > t0 + v + 8 * hi) s[sub][v] = -1e30f;

    // Online softmax: row reductions across the 16-lane group (masks 1..8).
    float corr[8];
#pragma unroll
    for (int v = 0; v < 8; ++v) {
      float a = s[0][v], c2 = s[1][v];
      float mx = fmaxf(a, c2);
#pragma unroll
      for (int off2 = 1; off2 < 16; off2 <<= 1)
        mx = fmaxf(mx, __shfl_xor(mx, off2, 32));
      const float mnew = fmaxf(mi[v], mx);
      const float cr = exp2f((mi[v] - mnew) * LOG2E);
      const float pa = exp2f((a  - mnew) * LOG2E);
      const float pb = exp2f((c2 - mnew) * LOG2E);
      float rs = pa + pb;
#pragma unroll
      for (int off2 = 1; off2 < 16; off2 <<= 1)
        rs += __shfl_xor(rs, off2, 32);
      li[v] = li[v] * cr + rs;
      mi[v] = mnew;
      corr[v] = cr;
      const int row = v + 8 * hi;  // C-layout -> A-layout via LDS
      pMy[row * LDT + l16]      = (bf16)pa;
      pMy[row * LDT + 16 + l16] = (bf16)pb;
    }
#pragma unroll
    for (int nt = 0; nt < 8; ++nt)
#pragma unroll
      for (int v = 0; v < 8; ++v) yacc[nt][v] *= corr[v];

    // y += P(16x32) . V(32x128)
    v16bf pf = load_afrag(pMy, LDT, l16, hi);
#pragma unroll
    for (int nt = 0; nt < 8; ++nt) {
      v16bf bfrag = load_bfrag(&vLds[buf][nt * 16 * LDT], LDT, l16, hi);
      yacc[nt] = wmma_bf16(pf, bfrag, yacc[nt]);
    }

#ifdef USE_ASYNC_LDS
    if (more) wait_async();
#else
    if (more) {
      *(bf16x8*)&kLds[nbuf][kOff]     = gk0;
      *(bf16x8*)&kLds[nbuf][kOff + 8] = gk1;
      *(bf16x8*)&vLds[nbuf][vOff]     = gv0;
      *(bf16x8*)&vLds[nbuf][vOff + 8] = gv1;
    }
#endif
    __syncthreads();
    buf = nbuf;
  }

  // Normalize and write y (B,T,C) bf16.
#pragma unroll
  for (int nt = 0; nt < 8; ++nt) {
#pragma unroll
    for (int v = 0; v < 8; ++v) {
      const float val = yacc[nt][v] / li[v];
      const int t = t0 + v + 8 * hi;
      const int col = (int)(h * kHD) + nt * 16 + l16;
      y[((size_t)b * kT + t) * kC + col] = (bf16)val;
    }
  }
}

// ---------------------------------------------------------------------------
// Kernel 5: output projection GEMM -> fp32 d_out, same tiling as kernel 3.
// ---------------------------------------------------------------------------
__global__ __launch_bounds__(256) void proj_gemm_kernel(
    const bf16* __restrict__ yb,     // (B*T, C)
    const bf16* __restrict__ wT,     // (C, C) N-major
    const float* __restrict__ bias,  // (C)
    float* __restrict__ out) {       // (B*T, C) fp32
  __shared__ __align__(16) bf16 aLds[2][128 * LDT];
  __shared__ __align__(16) bf16 bLds[2][128 * LDT];

  const int tid  = threadIdx.x;
  const int wave = tid >> 5;
  const int lane = tid & 31;
  const int l16  = lane & 15;
  const int hi   = lane >> 4;
  const int waveM = wave & 3;
  const int waveN = wave >> 2;
  const int mBlk = blockIdx.y * 128;
  const int nBlk = blockIdx.x * 128;

  v8f acc[2][4];
#pragma unroll
  for (int mt = 0; mt < 2; ++mt)
#pragma unroll
    for (int nt = 0; nt < 4; ++nt) acc[mt][nt] = zero8();

  const int r   = tid >> 1;
  const int seg = (tid & 1) * 16;
  const int off = r * LDT + seg;
  const bf16* srcA = yb + (size_t)(mBlk + r) * kC + seg;
  const bf16* srcB = wT + (size_t)(nBlk + r) * kC + seg;

#ifdef USE_ASYNC_LDS
  cp16_async(&aLds[0][off], srcA);
  cp16_async(&aLds[0][off + 8], srcA + 8);
  cp16_async(&bLds[0][off], srcB);
  cp16_async(&bLds[0][off + 8], srcB + 8);
  wait_async();
#else
  *(bf16x8*)&aLds[0][off]     = *(const bf16x8*)(srcA);
  *(bf16x8*)&aLds[0][off + 8] = *(const bf16x8*)(srcA + 8);
  *(bf16x8*)&bLds[0][off]     = *(const bf16x8*)(srcB);
  *(bf16x8*)&bLds[0][off + 8] = *(const bf16x8*)(srcB + 8);
#endif
  __syncthreads();

  constexpr int S = kC / 32;
  int buf = 0;
  for (int s = 0; s < S; ++s) {
    const int nbuf = buf ^ 1;
    const bool more = (s + 1) < S;
#ifdef USE_ASYNC_LDS
    if (more) {
      const bf16* na = srcA + (s + 1) * 32;
      const bf16* nb = srcB + (s + 1) * 32;
      cp16_async(&aLds[nbuf][off], na);
      cp16_async(&aLds[nbuf][off + 8], na + 8);
      cp16_async(&bLds[nbuf][off], nb);
      cp16_async(&bLds[nbuf][off + 8], nb + 8);
    }
#else
    bf16x8 ga0, ga1, gb0, gb1;
    if (more) {
      const bf16* na = srcA + (s + 1) * 32;
      const bf16* nb = srcB + (s + 1) * 32;
      ga0 = *(const bf16x8*)(na);
      ga1 = *(const bf16x8*)(na + 8);
      gb0 = *(const bf16x8*)(nb);
      gb1 = *(const bf16x8*)(nb + 8);
    }
#endif
    v16bf af[2], bf_[4];
#pragma unroll
    for (int mt = 0; mt < 2; ++mt)
      af[mt] = load_afrag(&aLds[buf][(waveM * 32 + mt * 16) * LDT], LDT, l16, hi);
#pragma unroll
    for (int nt = 0; nt < 4; ++nt)
      bf_[nt] = load_bfrag(&bLds[buf][(waveN * 64 + nt * 16) * LDT], LDT, l16, hi);
#pragma unroll
    for (int mt = 0; mt < 2; ++mt)
#pragma unroll
      for (int nt = 0; nt < 4; ++nt)
        acc[mt][nt] = wmma_bf16(af[mt], bf_[nt], acc[mt][nt]);

#ifdef USE_ASYNC_LDS
    if (more) wait_async();
#else
    if (more) {
      *(bf16x8*)&aLds[nbuf][off]     = ga0;
      *(bf16x8*)&aLds[nbuf][off + 8] = ga1;
      *(bf16x8*)&bLds[nbuf][off]     = gb0;
      *(bf16x8*)&bLds[nbuf][off + 8] = gb1;
    }
#endif
    __syncthreads();
    buf = nbuf;
  }

#pragma unroll
  for (int mt = 0; mt < 2; ++mt)
#pragma unroll
    for (int nt = 0; nt < 4; ++nt)
#pragma unroll
      for (int v = 0; v < 8; ++v) {
        const uint32_t rowg = (uint32_t)(mBlk + waveM * 32 + mt * 16 + v + 8 * hi);
        const uint32_t colg = (uint32_t)(nBlk + waveN * 64 + nt * 16 + l16);
        out[(size_t)rowg * kC + colg] = acc[mt][nt][v] + bias[colg];
      }
}

// ---------------------------------------------------------------------------
// Host launcher.  Workspace (bf16 staging, ~112 MB):
//   xb, W_attn^T, W_proj^T, q, k, v^T, y
// ---------------------------------------------------------------------------
extern "C" void kernel_launch(void* const* d_in, const int* in_sizes, int n_in,
                              void* d_out, int out_size, void* d_ws, size_t ws_size,
                              hipStream_t stream) {
  const float* x      = (const float*)d_in[0];
  const float* W_attn = (const float*)d_in[1];
  const float* b_attn = (const float*)d_in[2];
  const float* W_proj = (const float*)d_in[3];
  const float* b_proj = (const float*)d_in[4];
  float* out = (float*)d_out;

  char* ws = (char*)d_ws;
  auto alloc = [&](size_t bytes) -> char* {
    char* p = ws;
    ws += (bytes + 255) & ~(size_t)255;
    return p;
  };
  const size_t nX = (size_t)kB * kT * kC;
  bf16* xb  = (bf16*)alloc(nX * 2);
  bf16* waT = (bf16*)alloc((size_t)kN3 * kC * 2);
  bf16* wpT = (bf16*)alloc((size_t)kC * kC * 2);
  bf16* qb  = (bf16*)alloc(nX * 2);
  bf16* kb  = (bf16*)alloc(nX * 2);
  bf16* vtb = (bf16*)alloc(nX * 2);
  bf16* yb  = (bf16*)alloc(nX * 2);

  f32_to_bf16_kernel<<<2048, 256, 0, stream>>>(x, xb, nX);
  transpose_to_bf16_kernel<<<dim3(kN3 / 32, kC / 32), 256, 0, stream>>>(W_attn, waT, kC, kN3);
  transpose_to_bf16_kernel<<<dim3(kC / 32, kC / 32), 256, 0, stream>>>(W_proj, wpT, kC, kC);
  qkv_gemm_kernel<<<dim3(kN3 / 128, (kB * kT) / 128), 256, 0, stream>>>(
      xb, waT, b_attn, qb, kb, vtb);
  flash_attn_kernel<<<dim3(kT / 128, kB * kNH), 256, 0, stream>>>(qb, kb, vtb, yb);
  proj_gemm_kernel<<<dim3(kC / 128, (kB * kT) / 128), 256, 0, stream>>>(
      yb, wpT, b_proj, out);
}
